// RingAttention_8701603741982
// MI455X (gfx1250) — matrix operand order repaired and checked by reading, not compiled, not run on hardware
//
#include <hip/hip_runtime.h>
#include <hip/hip_bf16.h>
#include <stdint.h>

typedef __attribute__((ext_vector_type(16))) __bf16 v16bf;
typedef __attribute__((ext_vector_type(8)))  __bf16 v8bf;
typedef __attribute__((ext_vector_type(8)))  float  v8f;

#define B_    2
#define S_    8192
#define H_    16
#define D_    64
#define WAVES 8

static __device__ __forceinline__ __bf16 f2bf(float f) { return (__bf16)f; }

static __device__ __forceinline__ v8f vzero8() {
  v8f z = {0.f, 0.f, 0.f, 0.f, 0.f, 0.f, 0.f, 0.f};
  return z;
}

static __device__ __forceinline__ v16bf cat8(v8bf lo, v8bf hi) {
  return __builtin_shufflevector(lo, hi, 0, 1, 2, 3, 4, 5, 6, 7,
                                 8, 9, 10, 11, 12, 13, 14, 15);
}

// butterfly step within a 16-lane DPP row: lane <- lane ^ MASK
template <int MASK>
static __device__ __forceinline__ float dpp_xor(float x) {
  int v = __builtin_bit_cast(int, x);
  int r = __builtin_amdgcn_update_dpp(0, v, 0x160 | MASK, 0xf, 0xf, true);
  return __builtin_bit_cast(float, r);
}

static __device__ __forceinline__ float row_max16(float x) {
  x = fmaxf(x, dpp_xor<1>(x));
  x = fmaxf(x, dpp_xor<2>(x));
  x = fmaxf(x, dpp_xor<4>(x));
  x = fmaxf(x, dpp_xor<8>(x));
  return x;
}

static __device__ __forceinline__ float row_sum16(float x) {
  x += dpp_xor<1>(x);
  x += dpp_xor<2>(x);
  x += dpp_xor<4>(x);
  x += dpp_xor<8>(x);
  return x;
}

__global__ __launch_bounds__(256, 1) void ring_attn_fwd(
    const float* __restrict__ q, const float* __restrict__ k,
    const float* __restrict__ v, float* __restrict__ out) {
  // Double-buffered block-shared KV chunk (bf16), padded rows for
  // conflict-free 16-lane fragment reads; 16B-aligned segments.
  __shared__ __bf16 Kb[2][32][72];        // [buf][key][d]   (pad 64->72)
  __shared__ __bf16 Vt[2][64][40];        // [buf][d][key]   (pad 32->40)
  __shared__ __bf16 Pst[WAVES][16][40];   // per-wave P staging (pad 32->40)

  const int tid  = threadIdx.x;
  const int lane = tid & 31;
  const int wave = tid >> 5;
  const int nqb  = S_ / 128;              // 128 query rows per block
  const int qb   = blockIdx.x % nqb;
  const int h    = (blockIdx.x / nqb) % H_;
  const int b    = blockIdx.x / (nqb * H_);
  const int qt    = qb * 8 + wave;        // this wave's 16-row query tile
  const int qbase = qt * 16;
  const int qEnd  = qbase + 15;
  const int row   = lane & 15;            // M row (A) / N col (B/C) index
  const int hi    = (lane >> 4) & 1;      // half-wave select
  const float scale = 0.125f;             // 1/sqrt(64)
  const size_t strideS = (size_t)H_ * D_;

  // cooperative KV chunk fill: 32 keys, bf16-convert once per block
  auto fill = [&](int bi, int kv0) {
    {   // K: [key][d] row-major; 8 threads x 8 contiguous d per key
      const int key = tid >> 3;
      const int seg = (tid & 7) * 8;
      const float* src = k + (((size_t)b * S_ + kv0 + key) * H_ + h) * D_ + seg;
      v8bf t;
#pragma unroll
      for (int j = 0; j < 8; ++j) t[j] = f2bf(src[j]);
      *(v8bf*)&Kb[bi][key][seg] = t;
    }
    {   // V: transposed [d][key]; thread owns (d, 8 keys)
      const int d  = tid >> 2;
      const int ks = (tid & 3) * 8;
      const float* src = v + (((size_t)b * S_ + kv0 + ks) * H_ + h) * D_ + d;
      v8bf t;
#pragma unroll
      for (int j = 0; j < 8; ++j) t[j] = f2bf(src[(size_t)j * strideS]);
      *(v8bf*)&Vt[bi][d][ks] = t;
    }
  };

  // ---- Q A-fragments (16x32 bf16, two d-halves) -------------------------
  const float* qrow = q + (((size_t)b * S_ + qbase + row) * H_ + h) * D_;
  v16bf qf[2];
#pragma unroll
  for (int dh = 0; dh < 2; ++dh) {
    v16bf t;
    const int base = dh * 32 + (hi ? 16 : 0);
#pragma unroll
    for (int j = 0; j < 8; ++j) {
      t[j]     = f2bf(qrow[base + j]);        // K = base..base+7
      t[8 + j] = f2bf(qrow[base + 8 + j]);    // K = base+8..base+15
    }
    qf[dh] = t;
  }

  // ---- online-softmax running state -------------------------------------
  float mrow[8], lrow[8];
  v8f acc[4];
#pragma unroll
  for (int r = 0; r < 8; ++r) { mrow[r] = -1e30f; lrow[r] = 0.f; }
#pragma unroll
  for (int g = 0; g < 4; ++g) acc[g] = vzero8();

  const int blockKvEnd = qb * 128 + 127;   // max query row handled by block
  const int nch = blockKvEnd / 32 + 1;     // 32-key chunks to stream

  fill(0, 0);                              // prologue fill of buffer 0

  for (int c = 0; c < nch; ++c) {
    const int kv0 = c * 32;
    const int cur = c & 1;
    __syncthreads();                       // fill(cur) complete; buf cur^1 free
    if (c + 1 < nch) fill(cur ^ 1, kv0 + 32);
    if (kv0 + 64 <= blockKvEnd) {          // gfx1250 global_prefetch_b8
      __builtin_prefetch(
          k + (((size_t)b * S_ + kv0 + 64 + (tid >> 3)) * H_ + h) * D_, 0, 3);
      __builtin_prefetch(
          v + (((size_t)b * S_ + kv0 + 64 + (tid >> 3)) * H_ + h) * D_, 0, 3);
    }
    if (kv0 > qEnd) continue;              // wave-uniform causal skip

    // ---- preload all K AND V fragments (16 ds_load_b128 as one clause) --
    const int d0 = hi ? 16 : 0;            // B: K striped by half-wave
    const int kk = hi ? 16 : 0;
    v16bf kf00 = cat8(*(const v8bf*)&Kb[cur][row][d0],
                      *(const v8bf*)&Kb[cur][row][d0 + 8]);
    v16bf kf01 = cat8(*(const v8bf*)&Kb[cur][row][d0 + 32],
                      *(const v8bf*)&Kb[cur][row][d0 + 40]);
    v16bf kf10 = cat8(*(const v8bf*)&Kb[cur][16 + row][d0],
                      *(const v8bf*)&Kb[cur][16 + row][d0 + 8]);
    v16bf kf11 = cat8(*(const v8bf*)&Kb[cur][16 + row][d0 + 32],
                      *(const v8bf*)&Kb[cur][16 + row][d0 + 40]);
    v16bf vB0  = cat8(*(const v8bf*)&Vt[cur][row][kk],
                      *(const v8bf*)&Vt[cur][row][kk + 8]);
    v16bf vB1  = cat8(*(const v8bf*)&Vt[cur][16 + row][kk],
                      *(const v8bf*)&Vt[cur][16 + row][kk + 8]);
    v16bf vB2  = cat8(*(const v8bf*)&Vt[cur][32 + row][kk],
                      *(const v8bf*)&Vt[cur][32 + row][kk + 8]);
    v16bf vB3  = cat8(*(const v8bf*)&Vt[cur][48 + row][kk],
                      *(const v8bf*)&Vt[cur][48 + row][kk + 8]);

    // ---- S = Q K^T : two 16x16 f32 tiles; dh-outer (A operand reuse) ----
    v8f sc[2];
    sc[0] = vzero8(); sc[1] = vzero8();
    sc[0] = __builtin_amdgcn_wmma_f32_16x16x32_bf16(
        false, qf[0], false, kf00, (short)0, sc[0], false, false);
    sc[1] = __builtin_amdgcn_wmma_f32_16x16x32_bf16(
        false, qf[0], false, kf10, (short)0, sc[1], false, false);
    sc[0] = __builtin_amdgcn_wmma_f32_16x16x32_bf16(
        false, qf[1], false, kf01, (short)0, sc[0], false, false);
    sc[1] = __builtin_amdgcn_wmma_f32_16x16x32_bf16(
        false, qf[1], false, kf11, (short)0, sc[1], false, false);

#if __has_builtin(__builtin_amdgcn_sched_group_barrier)
    // pipeline: 16 DS reads (K+V frags) first, then the 4 QK^T WMMAs
    __builtin_amdgcn_sched_group_barrier(0x100, 16, 0);  // DS reads
    __builtin_amdgcn_sched_group_barrier(0x008, 4, 0);   // WMMA
#endif

    // ---- causal mask + scale + online softmax merge ----------------------
#pragma unroll
    for (int r = 0; r < 8; ++r) {
      const int mr = r + hi * 8;           // query row in tile (C layout)
      const int qg = qbase + mr;
      float s0 = (kv0 + row      <= qg) ? sc[0][r] * scale : -1e30f;
      float s1 = (kv0 + 16 + row <= qg) ? sc[1][r] * scale : -1e30f;
      float rm    = row_max16(fmaxf(s0, s1));
      float mnew  = fmaxf(mrow[r], rm);
      float alpha = __expf(mrow[r] - mnew);
      float p0 = __expf(s0 - mnew);
      float p1 = __expf(s1 - mnew);
      float rs = row_sum16(p0 + p1);
      lrow[r] = lrow[r] * alpha + rs;
      mrow[r] = mnew;
#pragma unroll
      for (int g = 0; g < 4; ++g) acc[g][r] *= alpha;
      Pst[wave][mr][row]      = f2bf(p0);  // stage P row-major as bf16
      Pst[wave][mr][16 + row] = f2bf(p1);
    }
    asm volatile("s_wait_dscnt 0x0" ::: "memory");  // in-wave DS RAW fence

    // ---- P as 16x32 A fragment, then the PV WMMA chain -------------------
    const int base = hi ? 16 : 0;
    v16bf pA = cat8(*(const v8bf*)&Pst[wave][row][base],
                    *(const v8bf*)&Pst[wave][row][base + 8]);
    acc[0] = __builtin_amdgcn_wmma_f32_16x16x32_bf16(
        false, pA, false, vB0, (short)0, acc[0], false, false);
    acc[1] = __builtin_amdgcn_wmma_f32_16x16x32_bf16(
        false, pA, false, vB1, (short)0, acc[1], false, false);
    acc[2] = __builtin_amdgcn_wmma_f32_16x16x32_bf16(
        false, pA, false, vB2, (short)0, acc[2], false, false);
    acc[3] = __builtin_amdgcn_wmma_f32_16x16x32_bf16(
        false, pA, false, vB3, (short)0, acc[3], false, false);
  }

  // ---- epilogue: normalize, store ----------------------------------------
#pragma unroll
  for (int r = 0; r < 8; ++r) {
    const int mr  = r + hi * 8;
    const float inv = 1.0f / lrow[r];
    float* orow = out + (((size_t)b * S_ + qbase + mr) * H_ + h) * D_ + row;
#pragma unroll
    for (int g = 0; g < 4; ++g) orow[g * 16] = acc[g][r] * inv;
  }
}

extern "C" void kernel_launch(void* const* d_in, const int* in_sizes, int n_in,
                              void* d_out, int out_size, void* d_ws, size_t ws_size,
                              hipStream_t stream) {
  (void)in_sizes; (void)n_in; (void)out_size; (void)d_ws; (void)ws_size;
  const float* q = (const float*)d_in[0];
  const float* k = (const float*)d_in[1];
  const float* v = (const float*)d_in[2];
  float* out = (float*)d_out;
  const int nBlocks = B_ * H_ * (S_ / 128);        // 2048 blocks
  dim3 grid(nBlocks), block(32 * WAVES);           // 256 threads = 8 waves
  ring_attn_fwd<<<grid, block, 0, stream>>>(q, k, v, out);
}
